// DAHHConv_11639361372552
// MI455X (gfx1250) — matrix operand order repaired and checked by reading, not compiled, not run on hardware
//
#include <hip/hip_runtime.h>

// MI455X / gfx1250, wave32.
// node_ft = (H @ ((H^T @ (x@theta)) / colsum(H))) / rowsum(H)
// N=20000, E=10000, IN=128, OUT=64. H = 800MB f32, read once per GEMM
// (1.6GB -> ~69us floor at 23.3 TB/s). H converted to f16 in-register via
// v_cvt_pk_rtz_f16_f32, math on v_wmma_f32_16x16x32_f16, degree sums fused
// as an extra ones-WMMA. H^T GEMM is K-split x2 (250 WGs) with deterministic
// f32 partials + a small combine kernel.

#define NN 20000
#define NE 10000
#define IC 128
#define OC 64

typedef __attribute__((ext_vector_type(16))) _Float16     v16h;
typedef __attribute__((ext_vector_type(8)))  float        v8f;
typedef __attribute__((ext_vector_type(4)))  float        v4f;
typedef __attribute__((ext_vector_type(4)))  unsigned int v4u;

constexpr int KSTEP = 32;   // K per wmma_f32_16x16x32_f16
constexpr int KPAD  = 40;   // f16 per LDS row (32 + 8 pad to spread banks)
constexpr int NSPLIT0 = 9984; // 312*32 ; second split gets 313*32 = 10016

// one v_cvt_pk_rtz_f16_f32: two f32 -> packed 2 x f16
__device__ __forceinline__ unsigned int pk2(float a, float b) {
  return __builtin_bit_cast(unsigned int, __builtin_amdgcn_cvt_pkrtz(a, b));
}

// ---------------- Kernel 0: yT[c][n] = (x @ theta)[n][c] in f16 -------------
__global__ __launch_bounds__(256) void xtheta_kernel(
    const float* __restrict__ x, const float* __restrict__ theta,
    _Float16* __restrict__ yT) {
  const int n = blockIdx.x * 256 + threadIdx.x;
  if (n >= NN) return;
  float acc[OC];
#pragma unroll
  for (int c = 0; c < OC; ++c) acc[c] = 0.f;
  for (int kc = 0; kc < IC; kc += 32) {   // x row read once; theta uniform -> scalar loads
    float xr[32];
#pragma unroll
    for (int i = 0; i < 32; i += 4) {
      v4f v = *(const v4f*)(x + (size_t)n * IC + kc + i);
      xr[i] = v[0]; xr[i + 1] = v[1]; xr[i + 2] = v[2]; xr[i + 3] = v[3];
    }
#pragma unroll 4
    for (int c = 0; c < OC; ++c) {
      float s = acc[c];
#pragma unroll
      for (int k = 0; k < 32; ++k) s += xr[k] * theta[(size_t)(kc + k) * OC + c];
      acc[c] = s;
    }
  }
#pragma unroll
  for (int c = 0; c < OC; ++c) yT[(size_t)c * NN + n] = (_Float16)acc[c];
}

// ---- Kernel 1: partial (H^T @ y) and partial de over a K split ------------
// grid (125, 2), block 160 = 5 waves; 80 e rows per block (80 | 10000).
__global__ __launch_bounds__(160) void edge_partial_kernel(
    const float* __restrict__ H, const _Float16* __restrict__ yT,
    float* __restrict__ pAcc /*[2][NE][OC]*/, float* __restrict__ pOnes /*[2][NE]*/) {
  __shared__ __align__(16) unsigned short ldsA[80 * KPAD]; // A=[e_local][k=n]
  __shared__ __align__(16) unsigned short ldsB[64 * KPAD]; // B=[c][k=n]
  const int tid = threadIdx.x, wave = tid >> 5, lane = tid & 31;
  const int hl = lane >> 4, l16 = lane & 15;
  const int e0 = blockIdx.x * 80;
  const int s  = blockIdx.y;
  const int nBeg = s ? NSPLIT0 : 0;
  const int nEnd = s ? NN : NSPLIT0;
  const int eg = tid % 20, q = tid / 20;   // A stage: 20 e-groups x 8 k-quads

  v8f acc[4] = {}; v8f accOnes = {};
  v16h ones;
#pragma unroll
  for (int i = 0; i < 16; ++i) ones[i] = (_Float16)1.0f;

  for (int n0 = nBeg; n0 < nEnd; n0 += KSTEP) {
    // stage A: H[n0:n0+32][e0:e0+80] -> f16, transposed to [e][k]
    {
      v4f f[4];
#pragma unroll
      for (int r = 0; r < 4; ++r)
        f[r] = __builtin_nontemporal_load(
            (const v4f*)(H + (size_t)(n0 + 4 * q + r) * NE + e0 + eg * 4));
#pragma unroll
      for (int i = 0; i < 4; ++i) {        // pack 4 consecutive k as b64
        unsigned long long pk =
            (unsigned long long)pk2(f[0][i], f[1][i]) |
            ((unsigned long long)pk2(f[2][i], f[3][i]) << 32);
        *(unsigned long long*)&ldsA[(eg * 4 + i) * KPAD + 4 * q] = pk;
      }
    }
    // stage B: yT[c][n0:n0+32] (already f16, K-contiguous)
    if (tid < 128) {
      const int c = tid >> 1, p = tid & 1;
      const _Float16* sp = yT + (size_t)c * NN + n0 + p * 16;
      *(v4u*)&ldsB[c * KPAD + p * 16]     = *(const v4u*)sp;
      *(v4u*)&ldsB[c * KPAD + p * 16 + 8] = *(const v4u*)(sp + 8);
    }
    __syncthreads();

    union { v4u u[2]; v16h h; } fa, fb;
    const int rowA = wave * 16 + l16, kbA = hl ? 8 : 0;     // ISA 16-bit A layout
    fa.u[0] = *(const v4u*)&ldsA[rowA * KPAD + kbA];
    fa.u[1] = *(const v4u*)&ldsA[rowA * KPAD + kbA + 16];
    const int kbB = hl ? 8 : 0;                             // ISA 16-bit B layout
#pragma unroll
    for (int t = 0; t < 4; ++t) {
      const int rowB = t * 16 + l16;
      fb.u[0] = *(const v4u*)&ldsB[rowB * KPAD + kbB];
      fb.u[1] = *(const v4u*)&ldsB[rowB * KPAD + kbB + 16];
      acc[t] = __builtin_amdgcn_wmma_f32_16x16x32_f16(
          false, fa.h, false, fb.h, (short)0, acc[t], false, false);
    }
    accOnes = __builtin_amdgcn_wmma_f32_16x16x32_f16(        // fused partial de
        false, fa.h, false, ones, (short)0, accOnes, false, false);
    __syncthreads();
  }

  const int eBase = e0 + wave * 16 + hl * 8;
#pragma unroll
  for (int t = 0; t < 4; ++t) {
    const int c = t * 16 + l16;            // lanes 0..15 -> contiguous 64B rows
#pragma unroll
    for (int j = 0; j < 8; ++j)
      pAcc[((size_t)s * NE + eBase + j) * OC + c] = acc[t][j];
  }
  if (l16 == 0) {                          // de replicated across cols: 1 lane writes
#pragma unroll
    for (int j = 0; j < 8; ++j) pOnes[(size_t)s * NE + eBase + j] = accOnes[j];
  }
}

// ---- Kernel 1b: combine splits, divide by de, emit edgeT f16 ---------------
__global__ __launch_bounds__(256) void edge_combine_kernel(
    const float* __restrict__ pAcc, const float* __restrict__ pOnes,
    _Float16* __restrict__ edgeT) {
  const int idx = blockIdx.x * 256 + threadIdx.x;   // 64*10000 exact
  const int c = idx / NE, e = idx - c * NE;
  const float num = pAcc[(size_t)e * OC + c] + pAcc[(size_t)(NE + e) * OC + c];
  const float den = pOnes[e] + pOnes[NE + e];
  edgeT[(size_t)c * NE + e] = (_Float16)(num / den);  // coalesced in e
}

// ------- Kernel 2: out[n][c] = (H @ edge)[n][c] / dn[n]   (f32 out) ---------
// Block 160 = 5 waves, 80 n rows (80 | 20000); K over E, zero-filled tail.
__global__ __launch_bounds__(160) void node_kernel(
    const float* __restrict__ H, const _Float16* __restrict__ edgeT,
    float* __restrict__ out) {
  __shared__ __align__(16) unsigned short ldsA[80 * KPAD]; // A=[m=n][k=e]
  __shared__ __align__(16) unsigned short ldsB[64 * KPAD]; // B=[c][k=e]
  const int tid = threadIdx.x, wave = tid >> 5, lane = tid & 31;
  const int hl = lane >> 4, l16 = lane & 15;
  const int n0 = blockIdx.x * 80;
  const int m = tid >> 1, part = tid & 1;  // A stage: 80 rows x 2 halves

  v8f acc[4] = {}; v8f accOnes = {};
  v16h ones;
#pragma unroll
  for (int i = 0; i < 16; ++i) ones[i] = (_Float16)1.0f;

  for (int e0 = 0; e0 < NE; e0 += KSTEP) {
    // stage A: H[n0+m][e0+k], K-contiguous, no transpose
#pragma unroll
    for (int i = 0; i < 4; ++i) {
      const int k = part * 16 + i * 4;
      v4f f = {0.f, 0.f, 0.f, 0.f};
      if (e0 + k < NE)
        f = __builtin_nontemporal_load(
            (const v4f*)(H + (size_t)(n0 + m) * NE + e0 + k));
      unsigned long long pk =
          (unsigned long long)pk2(f[0], f[1]) |
          ((unsigned long long)pk2(f[2], f[3]) << 32);
      *(unsigned long long*)&ldsA[m * KPAD + k] = pk;
    }
    // stage B: edgeT[c][e0:e0+32], zero-filled past E
    if (tid < 128) {
      const int c = tid >> 1, p = tid & 1;
      v4u b0 = {0u, 0u, 0u, 0u}, b1 = {0u, 0u, 0u, 0u};
      if (e0 + p * 16 < NE) {
        const _Float16* sp = edgeT + (size_t)c * NE + e0 + p * 16;
        b0 = *(const v4u*)sp; b1 = *(const v4u*)(sp + 8);
      }
      *(v4u*)&ldsB[c * KPAD + p * 16]     = b0;
      *(v4u*)&ldsB[c * KPAD + p * 16 + 8] = b1;
    }
    __syncthreads();

    union { v4u u[2]; v16h h; } fa, fb;
    const int rowA = wave * 16 + l16, kbA = hl ? 8 : 0;
    fa.u[0] = *(const v4u*)&ldsA[rowA * KPAD + kbA];
    fa.u[1] = *(const v4u*)&ldsA[rowA * KPAD + kbA + 16];
    const int kbB = hl ? 8 : 0;
#pragma unroll
    for (int t = 0; t < 4; ++t) {
      const int rowB = t * 16 + l16;
      fb.u[0] = *(const v4u*)&ldsB[rowB * KPAD + kbB];
      fb.u[1] = *(const v4u*)&ldsB[rowB * KPAD + kbB + 16];
      acc[t] = __builtin_amdgcn_wmma_f32_16x16x32_f16(
          false, fa.h, false, fb.h, (short)0, acc[t], false, false);
    }
    accOnes = __builtin_amdgcn_wmma_f32_16x16x32_f16(        // fused dn[n]
        false, fa.h, false, ones, (short)0, accOnes, false, false);
    __syncthreads();
  }

#pragma unroll
  for (int t = 0; t < 4; ++t) {
    const int c = t * 16 + l16;            // lanes 0..15 -> 64B contiguous rows
#pragma unroll
    for (int j = 0; j < 8; ++j) {
      const int n = n0 + wave * 16 + hl * 8 + j;
      __builtin_nontemporal_store(acc[t][j] / accOnes[j],
                                  out + (size_t)n * OC + c);
    }
  }
}

extern "C" void kernel_launch(void* const* d_in, const int* in_sizes, int n_in,
                              void* d_out, int out_size, void* d_ws, size_t ws_size,
                              hipStream_t stream) {
  const float* x     = (const float*)d_in[0];
  const float* H     = (const float*)d_in[1];
  const float* theta = (const float*)d_in[2];
  float* out = (float*)d_out;

  // workspace layout (bytes):
  //   yT    [64][20000] f16 : 0        .. 2,560,000
  //   edgeT [64][10000] f16 : 2,560,000 .. 3,840,000
  //   pAcc  [2][10000][64] f32 : 3,840,000 .. 8,960,000
  //   pOnes [2][10000]     f32 : 8,960,000 .. 9,040,000
  char* base = (char*)d_ws;
  _Float16* yT    = (_Float16*)(base);
  _Float16* edgeT = (_Float16*)(base + 2560000);
  float*    pAcc  = (float*)(base + 3840000);
  float*    pOnes = (float*)(base + 8960000);

  xtheta_kernel<<<(NN + 255) / 256, 256, 0, stream>>>(x, theta, yT);
  edge_partial_kernel<<<dim3(NE / 80, 2), 160, 0, stream>>>(H, yT, pAcc, pOnes);
  edge_combine_kernel<<<(NE * OC) / 256, 256, 0, stream>>>(pAcc, pOnes, edgeT);
  node_kernel<<<NN / 80, 160, 0, stream>>>(H, edgeT, out);
}
